// UnigramLM_36034775613665
// MI455X (gfx1250) — 8-bit matrix operand order repaired and checked by reading, compiled, not run on hardware
//
#include <hip/hip_runtime.h>
#include <hip/hip_bf16.h>

typedef __attribute__((ext_vector_type(8))) int v8i;

constexpr int   kT   = 2048;
constexpr int   kV   = 8192;
constexpr float kNEG = -1000000000.0f;

constexpr size_t ua256(size_t x) { return (x + 255) & ~(size_t)255; }
constexpr size_t SROW_OFF  = 0;
constexpr size_t SCOL_OFF  = ua256(SROW_OFF  + sizeof(double) * kT * 8);
constexpr size_t AD_OFF    = ua256(SCOL_OFF  + sizeof(double) * 8 * (kT + 8));
constexpr size_t CA_OFF    = ua256(AD_OFF    + sizeof(double) * (kT + 1));
constexpr size_t BD_OFF    = ua256(CA_OFF    + sizeof(float)  * (kT + 1));
constexpr size_t CB_OFF    = ua256(BD_OFF    + sizeof(double) * (kT + 1));
constexpr size_t ALPHA_OFF = ua256(CB_OFF    + sizeof(float)  * (kT + 1));
constexpr size_t BETA_OFF  = ua256(ALPHA_OFF + sizeof(float)  * (kT + 1));
constexpr size_t NORM_OFF  = ua256(BETA_OFF  + sizeof(float)  * (kT + 1));

__device__ __forceinline__ int dhi32(double x) {
  return (int)(__double_as_longlong(x) >> 32);
}
__device__ __forceinline__ double mk2pow(int e) {
  return __longlong_as_double(((long long)(1023 + e)) << 52);
}

__global__ __launch_bounds__(512) void ulm_s_kernel(
    const int* __restrict__ seq, const int* __restrict__ pieces,
    const int* __restrict__ piece_lens, const float* __restrict__ log_probs,
    double* __restrict__ Srow, double* __restrict__ Scolp) {
  __shared__ unsigned long long wk[32];
  __shared__ float sred[16][32][8];
  const int tid = threadIdx.x;
  const int t0  = blockIdx.x * 32;

  if (tid < 32) {
    unsigned long long k = 0;
    #pragma unroll
    for (int j = 0; j < 8; ++j) {
      int idx = t0 + tid + j;
      unsigned long long b =
          (idx < kT) ? (unsigned long long)((unsigned)seq[idx] & 0xFFu) : 0xFFull;
      k |= b << (8 * j);
    }
    wk[tid] = k;
  }
  __syncthreads();

  const int wave = tid >> 5;
  const int lane = tid & 31;
  const unsigned long long myw = wk[lane];

  float acc[8];
  #pragma unroll
  for (int l = 0; l < 8; ++l) acc[l] = 0.0f;

  const int vbeg = wave * (kV / 16);
  const int vend = vbeg + (kV / 16);
  for (int v = vbeg; v < vend; ++v) {
    int len = piece_lens[v];
    unsigned long long pk = 0;
    #pragma unroll
    for (int j = 0; j < 8; ++j)
      pk |= ((unsigned long long)((unsigned)pieces[v * 8 + j] & 0xFFu)) << (8 * j);
    unsigned long long mask = (len >= 8) ? ~0ull : ((1ull << (8 * len)) - 1ull);
    if (((myw ^ pk) & mask) == 0ull) {
      float pe = expf(log_probs[v]);
      #pragma unroll
      for (int l = 0; l < 8; ++l)
        if (len == l + 1) acc[l] += pe;
    }
  }

  #pragma unroll
  for (int l = 0; l < 8; ++l) sred[wave][lane][l] = acc[l];
  __syncthreads();

  if (tid < 256) {
    int tt = tid >> 3, l = tid & 7;
    float s = 0.0f;
    #pragma unroll
    for (int w = 0; w < 16; ++w) s += sred[w][tt][l];
    double sv = (double)s;
    int t = t0 + tt;
    Srow[(size_t)t * 8 + l]                 = sv;
    Scolp[(size_t)l * (kT + 8) + 8 + t]     = sv;
  }
  if (blockIdx.x == 0 && tid < 64)
    Scolp[(size_t)(tid >> 3) * (kT + 8) + (tid & 7)] = 0.0;
}

__global__ void ulm_scan_kernel(const double* __restrict__ Srow,
                                const double* __restrict__ Scolp,
                                double* __restrict__ Ad, float* __restrict__ Ca,
                                double* __restrict__ Bd, float* __restrict__ Cb) {
  if (threadIdx.x != 0) return;
  if (blockIdx.x == 0) {
    double F[8] = {0, 0, 0, 0, 0, 0, 0, 0};
    double A = 1.0, C = 0.0;
    Ad[0] = 1.0; Ca[0] = 0.0f;
    for (int t = 0; t < kT; t += 8) {
      double mx = A;
      #pragma unroll
      for (int j = 0; j < 8; ++j) mx = fmax(mx, F[j]);
      if (mx > 0.0) {
        int e = ((dhi32(mx) >> 20) & 0x7FF) - 1023;
        if (e < -64 || e > 64) {
          double s = mk2pow(-e);
          A *= s;
          #pragma unroll
          for (int j = 0; j < 8; ++j) F[j] *= s;
          C += (double)e * 0.6931471805599453;
        }
      }
      #pragma unroll
      for (int k = 0; k < 8; ++k) {
        const double* row = Srow + (size_t)(t + k) * 8;
        #pragma unroll
        for (int j = 0; j < 8; ++j) F[(k + j) & 7] += A * row[j];
        A = F[k];
        F[k] = 0.0;
        Ad[t + k + 1] = A;
        Ca[t + k + 1] = (float)C;
      }
    }
  } else {
    double F[8] = {0, 0, 0, 0, 0, 0, 0, 0};
    double B = 1.0, C = 0.0;
    Bd[kT] = 1.0; Cb[kT] = 0.0f;
    for (int ms = 0; ms < kT; ms += 8) {
      double mx = B;
      #pragma unroll
      for (int j = 0; j < 8; ++j) mx = fmax(mx, F[j]);
      if (mx > 0.0) {
        int e = ((dhi32(mx) >> 20) & 0x7FF) - 1023;
        if (e < -64 || e > 64) {
          double s = mk2pow(-e);
          B *= s;
          #pragma unroll
          for (int j = 0; j < 8; ++j) F[j] *= s;
          C += (double)e * 0.6931471805599453;
        }
      }
      #pragma unroll
      for (int k = 0; k < 8; ++k) {
        int m = ms + k;
        #pragma unroll
        for (int j = 0; j < 8; ++j)
          F[(k + j) & 7] += B * Scolp[(size_t)j * (kT + 8) + (kT + 7 - j) - m];
        B = F[k];
        F[k] = 0.0;
        Bd[kT - 1 - m] = B;
        Cb[kT - 1 - m] = (float)C;
      }
    }
  }
}

__global__ void ulm_finish_kernel(const double* __restrict__ Ad,
                                  const float* __restrict__ Ca,
                                  const double* __restrict__ Bd,
                                  const float* __restrict__ Cb,
                                  float* __restrict__ alphaF,
                                  float* __restrict__ betaF,
                                  float* __restrict__ normp) {
  int i = blockIdx.x * blockDim.x + threadIdx.x;
  if (i > kT) return;
  double a = Ad[i];
  float av = (a > 0.0) ? (float)(log(a) + (double)Ca[i]) : kNEG;
  alphaF[i] = av;
  double b = Bd[i];
  betaF[i] = (b > 0.0) ? (float)(log(b) + (double)Cb[i]) : kNEG;
  if (i == kT) normp[0] = av;
}

__global__ __launch_bounds__(256) void ulm_p_kernel(
    const int* __restrict__ seq, const int* __restrict__ pieces,
    const int* __restrict__ piece_lens, const float* __restrict__ log_probs,
    const float* __restrict__ alphaF, const float* __restrict__ betaF,
    const float* __restrict__ normp, float* __restrict__ out) {
  const int lane = threadIdx.x & 31;
  const int tile = blockIdx.x * (blockDim.x >> 5) + (threadIdx.x >> 5);
  const int tIdx = tile >> 9;
  const int vIdx = tile & 511;
  const int t0 = tIdx * 16, v0 = vIdx * 16;
  const int col = lane & 15;
  const int hi  = lane >> 4;

  const int v   = v0 + col;
  const int len = piece_lens[v];
  const float lp = log_probs[v];
  int4 p01 = *(const int4*)(pieces + (size_t)v * 8);
  int4 p23 = *(const int4*)(pieces + (size_t)v * 8 + 4);
  int pv[8] = {p01.x, p01.y, p01.z, p01.w, p23.x, p23.y, p23.z, p23.w};

  const int abase0[8] = {0, 4, 16, 20, 32, 36, 48, 52};
  v8i acc;
  #pragma unroll
  for (int i = 0; i < 8; ++i) acc[i] = 0;

  #pragma unroll
  for (int l = 0; l < 8; ++l) {
    int sidx = t0 + col + l;
    int s = (sidx < kT) ? seq[sidx] : -1;
    int pcol = (l < len) ? pv[l] : -1;
    v8i a, b;
    #pragma unroll
    for (int i = 0; i < 8; ++i) {
      int ab = abase0[i] + 8 * hi;
      unsigned d = (unsigned)(s - ab);
      a[i] = (d < 4u) ? (1 << ((int)(d & 3u) * 8)) : 0;
      int bb = abase0[i] + 8 * hi;
      unsigned e = (unsigned)(pcol - bb);
      b[i] = (e < 4u) ? (1 << ((int)(e & 3u) * 8)) : 0;
    }
    acc = __builtin_amdgcn_wmma_i32_16x16x64_iu8(false, a, false, b, acc,
                                                 false, false);
  }

  const float norm = normp[0];
  const bool ok = norm > (kNEG * 0.5f);
  const int tb = t0 + 8 * hi;
  float4 a0 = *(const float4*)(alphaF + tb);
  float4 a1 = *(const float4*)(alphaF + tb + 4);
  float avv[8] = {a0.x, a0.y, a0.z, a0.w, a1.x, a1.y, a1.z, a1.w};

  #pragma unroll
  for (int r = 0; r < 8; ++r) {
    int trow = tb + r;
    int bidx = trow + len; if (bidx > kT) bidx = kT;
    float bv = betaF[bidx];
    float pout = 0.0f;
    if (ok && acc[r] == len)
      pout = expf(avv[r] + lp + bv - norm);
    out[(size_t)trow * kV + v] = pout;
  }
}

extern "C" void kernel_launch(void* const* d_in, const int* in_sizes, int n_in,
                              void* d_out, int out_size, void* d_ws, size_t ws_size,
                              hipStream_t stream) {
  const int*   seq        = (const int*)d_in[0];
  const int*   pieces     = (const int*)d_in[1];
  const int*   piece_lens = (const int*)d_in[2];
  const float* log_probs  = (const float*)d_in[3];
  float* out = (float*)d_out;
  char*  ws  = (char*)d_ws;

  double* Srow   = (double*)(ws + SROW_OFF);
  double* Scolp  = (double*)(ws + SCOL_OFF);
  double* Ad     = (double*)(ws + AD_OFF);
  float*  Ca     = (float*) (ws + CA_OFF);
  double* Bd     = (double*)(ws + BD_OFF);
  float*  Cb     = (float*) (ws + CB_OFF);
  float*  alphaF = (float*) (ws + ALPHA_OFF);
  float*  betaF  = (float*) (ws + BETA_OFF);
  float*  normp  = (float*) (ws + NORM_OFF);

  ulm_s_kernel<<<kT / 32, 512, 0, stream>>>(seq, pieces, piece_lens, log_probs,
                                            Srow, Scolp);
  ulm_scan_kernel<<<2, 32, 0, stream>>>(Srow, Scolp, Ad, Ca, Bd, Cb);
  ulm_finish_kernel<<<(kT + 256) / 256, 256, 0, stream>>>(Ad, Ca, Bd, Cb,
                                                          alphaF, betaF, normp);
  ulm_p_kernel<<<(kT / 16) * (kV / 16) / 8, 256, 0, stream>>>(
      seq, pieces, piece_lens, log_probs, alphaF, betaF, normp, out);
}
